// TripletSemiHardLoss_69329362092398
// MI455X (gfx1250) — matrix operand order repaired and checked by reading, not compiled, not run on hardware
//
#include <hip/hip_runtime.h>
#include <hip/hip_bf16.h>

#define NN 512
#define DD 512
#define MARGIN 1.0f

typedef _Float16 v16h __attribute__((ext_vector_type(16)));
typedef _Float16 v8h  __attribute__((ext_vector_type(8)));
typedef float    v8f  __attribute__((ext_vector_type(8)));

// Kernel 1: fp32 -> fp16 conversion + row squared norms. One wave (32 lanes) per row.
__global__ void k_prep(const float* __restrict__ x, _Float16* __restrict__ xh,
                       float* __restrict__ sq) {
    int row  = (blockIdx.x * blockDim.x + threadIdx.x) >> 5;
    int lane = threadIdx.x & 31;
    if (row >= NN) return;
    const float* xr = x + row * DD;
    float s = 0.0f;
    #pragma unroll
    for (int t = 0; t < DD / 32; ++t) {
        float v = xr[lane + 32 * t];
        s += v * v;
        xh[row * DD + lane + 32 * t] = (_Float16)v;
    }
    #pragma unroll
    for (int off = 16; off > 0; off >>= 1) s += __shfl_xor(s, off, 32);
    if (lane == 0) sq[row] = s;
}

// Kernel 2: pd = max(sq_i + sq_j - 2 * (X X^T), 0), zero diagonal.
// One wave computes one 16x16 tile via v_wmma_f32_16x16x32_f16 (K-loop of 16 steps).
__global__ void k_gemm_pd(const _Float16* __restrict__ xh, const float* __restrict__ sq,
                          float* __restrict__ pd) {
    int w    = (blockIdx.x * blockDim.x + threadIdx.x) >> 5;   // tile id, 0..1023
    int lane = threadIdx.x & 31;
    int tr = w >> 5;        // tile row
    int tc = w & 31;        // tile col
    int m  = lane & 15;     // A row within tile / B col within tile
    int g  = lane >> 4;     // lane-half selector

    const _Float16* arow = xh + (tr * 16 + m) * DD;  // A: row tr*16+m of X
    const _Float16* brow = xh + (tc * 16 + m) * DD;  // B: col tc*16+m of X^T = row of X

    v8f c = {};
    #pragma unroll
    for (int kb = 0; kb < DD; kb += 32) {
        // A, contiguous k order chosen here (same as B): lane half g -> elems j = K(kb + 16g + j)
        v16h a = *(const v16h*)(arow + kb + g * 16);
        // B, contiguous k order chosen here (same as A): lane half g -> elems j = B[kb + 16g + j][m] = X[tc*16+m][kb+16g+j]
        v16h b = *(const v16h*)(brow + kb + g * 16);
        c = __builtin_amdgcn_wmma_f32_16x16x32_f16(false, a, false, b,
                                                   (short)0, c, false, false);
    }

    int j    = tc * 16 + m;       // column (C layout: N = lane % 16)
    float sj = sq[j];
    #pragma unroll
    for (int v = 0; v < 8; ++v) {
        int i = tr * 16 + v + 8 * g;   // row (C layout: M = vgpr + 8*(lane/16))
        float d2 = sq[i] + sj - 2.0f * c[v];
        d2 = fmaxf(d2, 0.0f);
        if (i == j) d2 = 0.0f;
        pd[i * NN + j] = d2;
    }
}

// Kernel 3: per-anchor-row semi-hard loss. One block (256 threads) per row r.
__global__ void k_loss(const float* __restrict__ pd, const int* __restrict__ lab,
                       float* __restrict__ rowLoss, float* __restrict__ rowPos) {
    __shared__ float spd[NN];
    __shared__ int   slab[NN];
    __shared__ float redf[256];
    __shared__ int   redi[256];

    int r = blockIdx.x, t = threadIdx.x;
    int k0 = t, k1 = t + 256;
    spd[k0] = pd[r * NN + k0];
    spd[k1] = pd[r * NN + k1];
    slab[k0] = lab[k0];
    slab[k1] = lab[k1];
    __syncthreads();

    int lr = slab[r];
    int n0 = (slab[k0] != lr);   // negative flags
    int n1 = (slab[k1] != lr);

    // rowmax
    redf[t] = fmaxf(spd[k0], spd[k1]);
    __syncthreads();
    for (int s = 128; s > 0; s >>= 1) {
        if (t < s) redf[t] = fmaxf(redf[t], redf[t + s]);
        __syncthreads();
    }
    float rowmax = redf[0];
    __syncthreads();

    // rowmin
    redf[t] = fminf(spd[k0], spd[k1]);
    __syncthreads();
    for (int s = 128; s > 0; s >>= 1) {
        if (t < s) redf[t] = fminf(redf[t], redf[t + s]);
        __syncthreads();
    }
    float rowmin = redf[0];
    __syncthreads();

    // neg_in = max_k((pd - rowmin) * adj_not) + rowmin  (masked-out entries contribute 0)
    float nloc = 0.0f;
    if (n0) nloc = fmaxf(nloc, spd[k0] - rowmin);
    if (n1) nloc = fmaxf(nloc, spd[k1] - rowmin);
    redf[t] = nloc;
    __syncthreads();
    for (int s = 128; s > 0; s >>= 1) {
        if (t < s) redf[t] = fmaxf(redf[t], redf[t + s]);
        __syncthreads();
    }
    float neg_in = redf[0] + rowmin;
    __syncthreads();

    // Loop over positives (same label, i != r); uniform branch on shared data.
    float lsum = 0.0f;
    int pcnt = 0;
    for (int i = 0; i < NN; ++i) {
        if (i == r || slab[i] != lr) continue;
        ++pcnt;
        float pdri = spd[i];
        // neg_out = min_k((pd - rowmax) * mask3) + rowmax ; masked-out contribute 0
        float mloc = 0.0f;
        int   floc = 0;
        if (n0 && spd[k0] > pdri) { mloc = fminf(mloc, spd[k0] - rowmax); floc = 1; }
        if (n1 && spd[k1] > pdri) { mloc = fminf(mloc, spd[k1] - rowmax); floc = 1; }
        redf[t] = mloc;
        redi[t] = floc;
        __syncthreads();
        for (int s = 128; s > 0; s >>= 1) {
            if (t < s) {
                redf[t] = fminf(redf[t], redf[t + s]);
                redi[t] |= redi[t + s];
            }
            __syncthreads();
        }
        float semi = redi[0] ? (redf[0] + rowmax) : neg_in;
        __syncthreads();
        lsum += fmaxf(MARGIN + pdri - semi, 0.0f);
    }
    if (t == 0) {
        rowLoss[r] = lsum;
        rowPos[r]  = (float)pcnt;
    }
}

// Kernel 4: deterministic fixed-order final reduction + division.
__global__ void k_final(const float* __restrict__ rowLoss, const float* __restrict__ rowPos,
                        float* __restrict__ out) {
    __shared__ float rl[256];
    __shared__ float rp[256];
    int t = threadIdx.x;
    rl[t] = rowLoss[t] + rowLoss[t + 256];
    rp[t] = rowPos[t] + rowPos[t + 256];
    __syncthreads();
    for (int s = 128; s > 0; s >>= 1) {
        if (t < s) { rl[t] += rl[t + s]; rp[t] += rp[t + s]; }
        __syncthreads();
    }
    if (t == 0) {
        float np = rp[0];
        if (np == 0.0f) np += 1e-5f;
        out[0] = rl[0] / np;
    }
}

extern "C" void kernel_launch(void* const* d_in, const int* in_sizes, int n_in,
                              void* d_out, int out_size, void* d_ws, size_t ws_size,
                              hipStream_t stream) {
    const float* x   = (const float*)d_in[0];   // [512, 512] fp32, L2-normalized rows
    const int*   tgt = (const int*)d_in[1];     // [512] int32 labels
    float* out = (float*)d_out;                 // scalar loss

    char* ws = (char*)d_ws;
    _Float16* xh = (_Float16*)ws;                                   // 512*512 f16 (512 KB)
    float* sq    = (float*)(ws + (size_t)NN * DD * 2);              // 512 f32
    float* pd    = (float*)(ws + (size_t)NN * DD * 2 + NN * 4);     // 512*512 f32 (1 MB)
    float* rowLoss = pd + (size_t)NN * NN;                          // 512 f32
    float* rowPos  = rowLoss + NN;                                  // 512 f32

    k_prep   <<<NN / 8, 256, 0, stream>>>(x, xh, sq);       // 8 waves/block, wave-per-row
    k_gemm_pd<<<(NN / 16) * (NN / 16) / 8, 256, 0, stream>>>(xh, sq, pd); // wave-per-tile
    k_loss   <<<NN, 256, 0, stream>>>(pd, tgt, rowLoss, rowPos);
    k_final  <<<1, 256, 0, stream>>>(rowLoss, rowPos, out);
}
